// Quantizer_85710367359200
// MI455X (gfx1250) — matrix operand order repaired and checked by reading, not compiled, not run on hardware
//
#include <hip/hip_runtime.h>
#include <hip/hip_bf16.h>
#include <math.h>

// ---------------------------------------------------------------------------
// Cosine-sim Gumbel quantizer for MI455X (gfx1250).
//  - GEMM in f16 WMMA (v_wmma_f32_16x16x32_f16), fp32 accum.
//  - Normalization done in fp32 before the f16 cast (unit vectors => good
//    relative precision in f16).
//  - Memory-bound on noise_u (537 MB of 805 MB total): streamed once with
//    TH=NT so the L2-resident codebook (regular-temporal) is not evicted.
// Workspace use: 1024*256*2 = 512 KB for the normalized f16 codebook.
// ---------------------------------------------------------------------------

typedef __attribute__((ext_vector_type(16))) _Float16 v16h;
typedef __attribute__((ext_vector_type(8)))  _Float16 v8h;
typedef __attribute__((ext_vector_type(8)))  float    v8f;
typedef __attribute__((ext_vector_type(4)))  float    v4f;

#define DDIM 256
#define KDIM 1024
#define EPSN 1e-8f

// ---------------------------------------------------------------------------
// Kernel 1: codebook row-normalize -> f16 (one block per codebook row).
// ---------------------------------------------------------------------------
__global__ __launch_bounds__(256) void cb_prep_kernel(const float* __restrict__ cb,
                                                      _Float16* __restrict__ cbh) {
  const int row = blockIdx.x;
  const int t   = threadIdx.x;
  float c = cb[(size_t)row * DDIM + t];
  float s = c * c;
#pragma unroll
  for (int m = 16; m >= 1; m >>= 1) s += __shfl_xor(s, m, 32);
  __shared__ float wsum[8];
  if ((t & 31) == 0) wsum[t >> 5] = s;
  __syncthreads();
  if (t == 0) {
    float tot = 0.f;
#pragma unroll
    for (int i = 0; i < 8; ++i) tot += wsum[i];
    wsum[0] = 1.0f / fmaxf(sqrtf(tot), EPSN);
  }
  __syncthreads();
  cbh[(size_t)row * DDIM + t] = (_Float16)(c * wsum[0]);
}

// ---------------------------------------------------------------------------
// Kernel 2: 16 rows of N per block; 8 waves; wave w covers K cols [128w,128w+128).
// ---------------------------------------------------------------------------
__global__ __launch_bounds__(256) void quant_kernel(const float* __restrict__ latent,
                                                    const float* __restrict__ noise,
                                                    const float* __restrict__ cb,
                                                    const _Float16* __restrict__ cbh,
                                                    const float* __restrict__ tptr,
                                                    float* __restrict__ out) {
  __shared__ __align__(16) _Float16 aLds[16 * DDIM];  // 8 KB normalized f16 latent tile
  __shared__ float partial[16][17];
  __shared__ float invLn[16];
  __shared__ float wBestV[8][16];
  __shared__ int   wBestI[8][16];
  __shared__ int   finalIdx[16];

  const int t       = threadIdx.x;
  const int rowBase = blockIdx.x * 16;

  // ---- Stage A: fp32 row norms, cast normalized rows to f16 in LDS --------
  const int m = t >> 4;       // row within tile (0..15)
  const int j = t & 15;       // 16-float chunk within row
  float av[16];
  {
    // latent is read exactly once across the whole dispatch -> NT loads.
    const v4f* lrow = (const v4f*)(latent + (size_t)(rowBase + m) * DDIM + j * 16);
    float s = 0.f;
#pragma unroll
    for (int q = 0; q < 4; ++q) {
      v4f x = __builtin_nontemporal_load(lrow + q);
#pragma unroll
      for (int i = 0; i < 4; ++i) { av[q * 4 + i] = x[i]; s += x[i] * x[i]; }
    }
    partial[m][j] = s;
  }
  __syncthreads();
  if (t < 16) {
    float tot = 0.f;
#pragma unroll
    for (int i = 0; i < 16; ++i) tot += partial[t][i];
    invLn[t] = 1.0f / fmaxf(sqrtf(tot), EPSN);
  }
  __syncthreads();
  {
    const float inv = invLn[m];
#pragma unroll
    for (int i = 0; i < 16; ++i)
      aLds[m * DDIM + j * 16 + i] = (_Float16)(av[i] * inv);
  }
  __syncthreads();

  // ---- Stage B: WMMA GEMM, 16x128 per wave, D=256 reduction ---------------
  const int wave    = t >> 5;
  const int lane    = t & 31;
  const int colBase = wave * 128;
  const int halfSel = lane >> 4;        // 0: lanes 0-15, 1: lanes 16-31
  const int mrow    = lane & 15;
  const int rHalf   = halfSel * 8;      // C-matrix rows owned by this half
  const int kbaseA  = halfSel * 8;      // unused; A frag also uses kbaseB
  const int kbaseB  = halfSel * 16;     // B-frag K striping (16-bit B 32x16)

  // Warm the noise lines this lane will need in the epilogue into GL2
  // (DEV-scope prefetch: skips WGP cache, fills GL2).
#pragma unroll
  for (int r = 0; r < 8; ++r)
    __builtin_prefetch(noise + (size_t)(rowBase + rHalf + r) * KDIM + colBase + mrow, 0, 1);

  v8f acc[8];
#pragma unroll
  for (int c = 0; c < 8; ++c) acc[c] = (v8f)0.0f;

  const _Float16* aRow = aLds + mrow * DDIM;
#pragma unroll
  for (int kt = 0; kt < 8; ++kt) {
    const int k0 = kt * 32;
    // A fragment: K = k0+kbaseB+{0..15} (contiguous, matches B)
    v8h alo = *(const v8h*)(aRow + k0 + kbaseB);
    v8h ahi = *(const v8h*)(aRow + k0 + kbaseB + 8);
    v16h af = __builtin_shufflevector(alo, ahi, 0, 1, 2, 3, 4, 5, 6, 7,
                                                8, 9, 10, 11, 12, 13, 14, 15);
#pragma unroll
    for (int c = 0; c < 8; ++c) {
      // B fragment: lane owns column (colBase+16c+mrow), 16 consecutive K.
      // Codebook is hot in L2 (reused by all 8192 blocks) -> regular loads.
      const _Float16* bRow = cbh + (size_t)(colBase + c * 16 + mrow) * DDIM + k0 + kbaseB;
      v8h blo = *(const v8h*)(bRow);
      v8h bhi = *(const v8h*)(bRow + 8);
      v16h bf = __builtin_shufflevector(blo, bhi, 0, 1, 2, 3, 4, 5, 6, 7,
                                                  8, 9, 10, 11, 12, 13, 14, 15);
      acc[c] = __builtin_amdgcn_wmma_f32_16x16x32_f16(
          /*neg_a=*/false, af, /*neg_b=*/false, bf,
          /*c_mod=*/(short)0, acc[c], /*reuse_a=*/false, /*reuse_b=*/false);
    }
  }

  // ---- Stage C: gumbel + argmax -------------------------------------------
  const float tinv = 1.0f / tptr[0];
  float bestV[8];
  int   bestI[8];
#pragma unroll
  for (int r = 0; r < 8; ++r) { bestV[r] = -3.402823466e38f; bestI[r] = 0; }

#pragma unroll
  for (int c = 0; c < 8; ++c) {
    const int col = colBase + c * 16 + mrow;
#pragma unroll
    for (int r = 0; r < 8; ++r) {
      const int row = rowBase + rHalf + r;
      // noise_u is the dominant stream (537 MB), read once -> NT load.
      const float u = __builtin_nontemporal_load(noise + (size_t)row * KDIM + col);
      const float g = -__logf(-__logf(u));
      const float v = acc[c][r] * tinv + g;
      if (v > bestV[r] || (v == bestV[r] && col < bestI[r])) { bestV[r] = v; bestI[r] = col; }
    }
  }
  // reduce across the 16 lanes that hold each row (xor within 16-lane halves)
#pragma unroll
  for (int msk = 8; msk >= 1; msk >>= 1) {
#pragma unroll
    for (int r = 0; r < 8; ++r) {
      const float ov = __shfl_xor(bestV[r], msk, 16);
      const int   oi = __shfl_xor(bestI[r], msk, 16);
      if (ov > bestV[r] || (ov == bestV[r] && oi < bestI[r])) { bestV[r] = ov; bestI[r] = oi; }
    }
  }
  if (mrow == 0) {
#pragma unroll
    for (int r = 0; r < 8; ++r) {
      wBestV[wave][rHalf + r] = bestV[r];
      wBestI[wave][rHalf + r] = bestI[r];
    }
  }
  __syncthreads();
  if (t < 16) {
    float bv = wBestV[0][t];
    int   bi = wBestI[0][t];
#pragma unroll
    for (int w2 = 1; w2 < 8; ++w2) {
      const float ov = wBestV[w2][t];
      const int   oi = wBestI[w2][t];
      if (ov > bv || (ov == bv && oi < bi)) { bv = ov; bi = oi; }
    }
    finalIdx[t] = bi;
  }
  __syncthreads();

  // ---- Stage D: gather winning codebook rows (fp32) -----------------------
  {
    const int idx = finalIdx[m];
    const v4f* src = (const v4f*)(cb + (size_t)idx * DDIM + j * 16);  // L2-hot: RT loads
    v4f* dst = (v4f*)(out + (size_t)(rowBase + m) * DDIM + j * 16);   // write-once: NT stores
#pragma unroll
    for (int i = 0; i < 4; ++i) {
      v4f x = src[i];
      __builtin_nontemporal_store(x, dst + i);
    }
  }
}

// ---------------------------------------------------------------------------
extern "C" void kernel_launch(void* const* d_in, const int* in_sizes, int n_in,
                              void* d_out, int out_size, void* d_ws, size_t ws_size,
                              hipStream_t stream) {
  const float* latent = (const float*)d_in[0];   // [N, 256] fp32
  const float* noise  = (const float*)d_in[1];   // [N, 1024] fp32
  const float* cb     = (const float*)d_in[2];   // [1024, 256] fp32
  const float* temp   = (const float*)d_in[3];   // scalar fp32
  float* out = (float*)d_out;                    // [N, 256] fp32

  _Float16* cbh = (_Float16*)d_ws;               // 512 KB normalized f16 codebook

  const int nRows = in_sizes[0] / DDIM;          // N = 131072

  cb_prep_kernel<<<KDIM, 256, 0, stream>>>(cb, cbh);
  quant_kernel<<<nRows / 16, 256, 0, stream>>>(latent, noise, cb, cbh, temp, out);
}
